// MPNNLayer_59167469469860
// MI455X (gfx1250) — matrix operand order repaired and checked by reading, not compiled, not run on hardware
//
#include <hip/hip_runtime.h>

#define E_EDGES 500000
#define FDIM 128
#define KDIM 256

typedef __attribute__((ext_vector_type(16))) __bf16 v16bf;
typedef __attribute__((ext_vector_type(8)))  __bf16 v8bf;
typedef __attribute__((ext_vector_type(8)))  float  v8f;
typedef __attribute__((ext_vector_type(4)))  float  v4f;
typedef __attribute__((ext_vector_type(4)))  unsigned int v4u;
typedef __attribute__((ext_vector_type(8)))  int v8i;
typedef __attribute__((ext_vector_type(4)))  int v4i;

#if defined(__has_builtin)
#if __has_builtin(__builtin_amdgcn_tensor_load_to_lds)
#define HAVE_TDM 1
#else
#define HAVE_TDM 0
#endif
#else
#define HAVE_TDM 0
#endif

// LDS layout in bf16 elements: W1T [128][264], W2T [128][136], H1 [8][16][136]
#define W1_STRIDE 264
#define W2_STRIDE 136
#define H1_STRIDE 136
#define SM_W1 0
#define SM_W2 (128 * W1_STRIDE)
#define SM_H1 (SM_W2 + 128 * W2_STRIDE)
#define SMEM_ELEMS (SM_H1 + 8 * 16 * H1_STRIDE)
#define SMEM_BYTES (SMEM_ELEMS * 2)

// -------- convert + transpose weights to bf16 in workspace --------
__global__ void wconvert_kernel(const float* __restrict__ W1,
                                const float* __restrict__ W2,
                                __bf16* __restrict__ w1t,
                                __bf16* __restrict__ w2t) {
  int idx = blockIdx.x * blockDim.x + threadIdx.x;
  if (idx < KDIM * FDIM) {               // W1 [256][128] -> w1t [128][256]
    int k = idx / FDIM, n = idx % FDIM;
    w1t[n * KDIM + k] = (__bf16)W1[idx];
  }
  if (idx < FDIM * FDIM) {               // W2 [128][128] -> w2t [128][128]
    int k = idx / FDIM, n = idx % FDIM;
    w2t[n * FDIM + k] = (__bf16)W2[idx];
  }
}

// xor-butterfly add within each 16-lane half (wave32, EXEC all-ones here)
template <int IMM>
__device__ __forceinline__ float swz(float v) {
  return __builtin_bit_cast(
      float, __builtin_amdgcn_ds_swizzle(__builtin_bit_cast(int, v), IMM));
}
__device__ __forceinline__ float redsum16(float v) {
  v += swz<0x041F>(v);  // xor 1
  v += swz<0x081F>(v);  // xor 2
  v += swz<0x101F>(v);  // xor 4
  v += swz<0x201F>(v);  // xor 8
  return v;
}

__device__ __forceinline__ v16bf cat8(v8bf lo, v8bf hi) {
  return __builtin_shufflevector(lo, hi, 0, 1, 2, 3, 4, 5, 6, 7,
                                 8, 9, 10, 11, 12, 13, 14, 15);
}

// -------- fused MPNN edge-MLP + LayerNorm --------
__global__ void __launch_bounds__(256, 1)
mpnn_fused_kernel(const float* __restrict__ in_right,
                  const float* __restrict__ in_left,
                  const __bf16* __restrict__ w1t,
                  const __bf16* __restrict__ w2t,
                  const float* __restrict__ b1, const float* __restrict__ b2,
                  const float* __restrict__ gamma,
                  const float* __restrict__ beta, float* __restrict__ out) {
  extern __shared__ __bf16 smem[];
  const int tid = threadIdx.x;
  const int wid = tid >> 5;
  const int lane = tid & 31;
  const int laneRow = lane & 15;  // A-matrix row / C-matrix column (mod 16)
  const int half = lane >> 4;
  const int c0 = half * 8;

#if HAVE_TDM
  // ---- stage bf16 weights into LDS via the Tensor Data Mover ----
  // TDM pad fields reproduce the padded LDS row strides:
  //   W1: 512B rows + 16B pad  -> 528B stride (264 bf16)
  //   W2: 256B rows + 16B pad  -> 272B stride (136 bf16)
  if (tid < 32) {  // one wave issues both descriptors; EXEC ignored by TDM
    const v4i gz = {0, 0, 0, 0};
    const v8i gz8 = {0, 0, 0, 0, 0, 0, 0, 0};
    {
      unsigned long long ga = (unsigned long long)(uintptr_t)w1t;
      v4u g0 = {1u,                                   // count=1, user D#
                (unsigned)(SM_W1 * 2),                // lds_addr (bytes)
                (unsigned)(ga & 0xFFFFFFFFu),         // global_addr[31:0]
                (unsigned)((ga >> 32) & 0x01FFFFFFu)  // global_addr[56:32]
                    | (2u << 30)};                    // type=2 (image)
      v8i g1 = {
          (int)((1u << 16)      // data_size = 2 bytes (bf16)
                | (1u << 20)    // pad_enable
                | (6u << 22)    // pad_interval: every 128 dwords (512B)
                | (3u << 25)),  // pad_amount: 4 dwords (16B)
          (int)(256u << 16),    // tensor_dim0 = 256 (low 16 in [63:48])
          (int)(128u << 16),    // tensor_dim0 hi = 0; tensor_dim1 = 128
          (int)(256u << 16),    // tensor_dim1 hi = 0; tile_dim0 = 256
          (int)(128u),          // tile_dim1 = 128; tile_dim2 = 0
          (int)(256),           // tensor_dim0_stride = 256 elements
          0, 0};                // dim1_stride unused for 2D tile
      __builtin_amdgcn_tensor_load_to_lds(g0, g1, gz, gz, gz8, 0);
    }
    {
      unsigned long long gb = (unsigned long long)(uintptr_t)w2t;
      v4u g0 = {1u, (unsigned)(SM_W2 * 2), (unsigned)(gb & 0xFFFFFFFFu),
                (unsigned)((gb >> 32) & 0x01FFFFFFu) | (2u << 30)};
      v8i g1 = {
          (int)((1u << 16) | (1u << 20)
                | (5u << 22)    // pad_interval: every 64 dwords (256B)
                | (3u << 25)),  // pad_amount: 4 dwords (16B)
          (int)(128u << 16),    // tensor_dim0 = 128
          (int)(128u << 16),    // tensor_dim1 = 128
          (int)(128u << 16),    // tile_dim0 = 128
          (int)(128u),          // tile_dim1 = 128
          (int)(128),           // tensor_dim0_stride = 128 elements
          0, 0};
      __builtin_amdgcn_tensor_load_to_lds(g0, g1, gz, gz, gz8, 0);
    }
    __builtin_amdgcn_s_wait_tensorcnt(0);
  }
  __syncthreads();
#else
  // ---- fallback: cooperative VALU copy of bf16 weights into LDS ----
  {
    const v4u* g1 = (const v4u*)w1t;
    v4u* l1 = (v4u*)(smem + SM_W1);
#pragma unroll
    for (int i = 0; i < 16; ++i) {
      int c = tid + 256 * i;
      int r = c >> 4, ch = c & 15;
      l1[r * (W1_STRIDE / 8) + ch] = g1[r * (KDIM / 8) + ch];
    }
    const v4u* g2 = (const v4u*)w2t;
    v4u* l2 = (v4u*)(smem + SM_W2);
#pragma unroll
    for (int i = 0; i < 8; ++i) {
      int c = tid + 256 * i;
      int r = c >> 3, ch = c & 7;
      l2[r * (W2_STRIDE / 8) + ch] = g2[r * (FDIM / 8) + ch];
    }
  }
  __syncthreads();
#endif

  const bool isRL = wid >= 4;                    // waves 0-3: l||r, 4-7: r||l
  const int eBase = blockIdx.x * 64 + (wid & 3) * 16;
  int eA = eBase + laneRow;
  if (eA >= E_EDGES) eA = E_EDGES - 1;           // clamp loads, guard stores
  const float* rowA = (isRL ? in_right : in_left) + (size_t)eA * FDIM;
  const float* rowB = (isRL ? in_left : in_right) + (size_t)eA * FDIM;

  float b1v[8], b2v[8], gmv[8], btv[8];
#pragma unroll
  for (int t = 0; t < 8; ++t) {
    int coln = 16 * t + laneRow;
    b1v[t] = b1[coln];
    b2v[t] = b2[coln];
    gmv[t] = gamma[coln];
    btv[t] = beta[coln];
  }

  // ---- GEMM1: 16x256 @ 256x128, bf16 WMMA, f32 accumulate ----
  v8f acc[8] = {};
  const __bf16* sW1 = smem + SM_W1;
#pragma unroll
  for (int ks = 0; ks < 8; ++ks) {
    int ca = 32 * ks + c0;   // 8-float chunks never straddle the 128 boundary
    int cb = ca + 16;
    const float* pa = (ca < FDIM) ? (rowA + ca) : (rowB + (ca - FDIM));
    const float* pb = (cb < FDIM) ? (rowA + cb) : (rowB + (cb - FDIM));
    v4f f0 = *(const v4f*)pa;
    v4f f1 = *(const v4f*)(pa + 4);
    v4f f2 = *(const v4f*)pb;
    v4f f3 = *(const v4f*)(pb + 4);
    v16bf a;
#pragma unroll
    for (int j = 0; j < 4; ++j) {
      a[j] = (__bf16)f0[j];
      a[4 + j] = (__bf16)f1[j];
      a[8 + j] = (__bf16)f2[j];
      a[12 + j] = (__bf16)f3[j];
    }
#pragma unroll
    for (int t = 0; t < 8; ++t) {
      const __bf16* bp =
          sW1 + (16 * t + laneRow) * W1_STRIDE + 32 * ks + 8 * half;
      v16bf b = cat8(*(const v8bf*)bp, *(const v8bf*)(bp + 16));
      acc[t] = __builtin_amdgcn_wmma_f32_16x16x32_bf16(false, a, false, b,
                                                       (short)0, acc[t],
                                                       false, false);
    }
  }

  // ---- bias + relu, stage h1 (bf16) in LDS for re-fragmenting ----
  __bf16* myH1 = smem + SM_H1 + wid * 16 * H1_STRIDE;
#pragma unroll
  for (int t = 0; t < 8; ++t) {
    int coln = 16 * t + laneRow;
#pragma unroll
    for (int v = 0; v < 8; ++v) {
      float x = acc[t][v] + b1v[t];
      x = fmaxf(x, 0.0f);
      myH1[(v + 8 * half) * H1_STRIDE + coln] = (__bf16)x;  // C-layout row
    }
  }
  asm volatile("s_wait_dscnt 0x0" ::: "memory");  // same-wave LDS RAW

  // ---- GEMM2: 16x128 @ 128x128 ----
  v8f acc2[8] = {};
  const __bf16* sW2 = smem + SM_W2;
#pragma unroll
  for (int ks = 0; ks < 4; ++ks) {
    const __bf16* ap = myH1 + laneRow * H1_STRIDE + 32 * ks + c0;
    v16bf a = cat8(*(const v8bf*)ap, *(const v8bf*)(ap + 16));
#pragma unroll
    for (int t = 0; t < 8; ++t) {
      const __bf16* bp =
          sW2 + (16 * t + laneRow) * W2_STRIDE + 32 * ks + 8 * half;
      v16bf b = cat8(*(const v8bf*)bp, *(const v8bf*)(bp + 16));
      acc2[t] = __builtin_amdgcn_wmma_f32_16x16x32_bf16(false, a, false, b,
                                                        (short)0, acc2[t],
                                                        false, false);
    }
  }

  // ---- bias + relu, then LayerNorm over the 128-wide row ----
#pragma unroll
  for (int t = 0; t < 8; ++t)
#pragma unroll
    for (int v = 0; v < 8; ++v) acc2[t][v] = fmaxf(acc2[t][v] + b2v[t], 0.0f);

#pragma unroll
  for (int v = 0; v < 8; ++v) {
    float s = 0.0f, q = 0.0f;
#pragma unroll
    for (int t = 0; t < 8; ++t) {
      float x = acc2[t][v];
      s += x;
      q += x * x;
    }
    s = redsum16(s);  // row lives across the 16 lanes of this half
    q = redsum16(q);
    float mu = s * (1.0f / 128.0f);
    float var = q * (1.0f / 128.0f) - mu * mu;
    float rstd = rsqrtf(var + 1e-5f);
    int eo = eBase + v + 8 * half;  // output row = edge index of this C row
    if (eo < E_EDGES) {
      float* orow = out + ((isRL ? (size_t)E_EDGES : (size_t)0) + eo) * FDIM;
#pragma unroll
      for (int t = 0; t < 8; ++t) {
        int coln = 16 * t + laneRow;
        orow[coln] = (acc2[t][v] - mu) * rstd * gmv[t] + btv[t];
      }
    }
  }
}

extern "C" void kernel_launch(void* const* d_in, const int* in_sizes, int n_in,
                              void* d_out, int out_size, void* d_ws,
                              size_t ws_size, hipStream_t stream) {
  (void)in_sizes; (void)n_in; (void)out_size; (void)ws_size;
  const float* in_right = (const float*)d_in[0];
  const float* in_left  = (const float*)d_in[1];
  const float* W1 = (const float*)d_in[2];
  const float* b1 = (const float*)d_in[3];
  const float* W2 = (const float*)d_in[4];
  const float* b2 = (const float*)d_in[5];
  const float* gamma = (const float*)d_in[6];
  const float* beta  = (const float*)d_in[7];
  float* out = (float*)d_out;

  __bf16* w1t = (__bf16*)d_ws;                          // 64 KB
  __bf16* w2t = (__bf16*)((char*)d_ws + 64 * 1024);     // 32 KB

  (void)hipFuncSetAttribute((const void*)mpnn_fused_kernel,
                            hipFuncAttributeMaxDynamicSharedMemorySize,
                            SMEM_BYTES);

  wconvert_kernel<<<(KDIM * FDIM + 255) / 256, 256, 0, stream>>>(W1, W2, w1t,
                                                                 w2t);
  mpnn_fused_kernel<<<(E_EDGES + 63) / 64, 256, SMEM_BYTES, stream>>>(
      in_right, in_left, w1t, w2t, b1, b2, gamma, beta, out);
}
